// GroupedQueryAttention_1898375544916
// MI455X (gfx1250) — matrix operand order repaired and checked by reading, not compiled, not run on hardware
//
#include <hip/hip_runtime.h>
#include <hip/hip_bf16.h>
#include <cstdint>

// ---------------------------------------------------------------------------
// Types for gfx1250 WMMA (wave32): v_wmma_f32_16x16x32_bf16
// ---------------------------------------------------------------------------
typedef __attribute__((ext_vector_type(16))) __bf16 v16bf;
typedef __attribute__((ext_vector_type(8)))  __bf16 v8bf;
typedef __attribute__((ext_vector_type(8)))  float  v8f;
typedef __attribute__((ext_vector_type(4)))  unsigned int v4u;
typedef __attribute__((ext_vector_type(8)))  int v8i_;
typedef __attribute__((ext_vector_type(4)))  int v4i_;

#define B_   4
#define S_   2048
#define D_   1024
#define HQ_  16
#define HKV_ 4
#define HD_  64
#define REP_ 4

__device__ __forceinline__ unsigned rfl(unsigned x) {
  return (unsigned)__builtin_amdgcn_readfirstlane((int)x);
}

// ---------------------------------------------------------------------------
// Tensor Data Mover: build a D# per CDNA5 ISA (CDNA5 ISA) and
// issue TENSOR_LOAD_TO_LDS. 1D/2D tiles, bf16 (data_size=2B).
//   group0: [1:0]=count=1 | lds_addr | global_addr[56:0] | type=2
//   group1: data_size=1(2B) | tensor_dim0/1 | tile_dim0/1 | dim0_stride
// All descriptor dwords pushed to SGPRs via readfirstlane.
// This toolchain exposes the 6-argument builtin form.
// ---------------------------------------------------------------------------
__device__ __forceinline__ void tdm_load_to_lds(
    unsigned lds_byte_off, const void* gptr,
    unsigned tile0, unsigned tile1,          // tile dims (elements); tile1=0 -> 1D
    unsigned tens0, unsigned tens1,          // tensor dims (elements)
    unsigned stride0) {                      // dim0 stride (elements)
  uint64_t ga = (uint64_t)(uintptr_t)gptr;
  v4u g0;
  g0[0] = rfl(1u);                                   // count=1, user descriptor
  g0[1] = rfl(lds_byte_off);
  g0[2] = rfl((unsigned)ga);
  g0[3] = rfl(((unsigned)(ga >> 32) & 0x01FFFFFFu) | (2u << 30)); // type=2
  v8i_ g1;
  g1[0] = (int)rfl(0x00010000u);                     // data_size=1 -> 2 bytes
  g1[1] = (int)rfl((tens0 & 0xFFFFu) << 16);         // tensor_dim0[15:0]
  g1[2] = (int)rfl(((tens0 >> 16) & 0xFFFFu) | ((tens1 & 0xFFFFu) << 16));
  g1[3] = (int)rfl(((tens1 >> 16) & 0xFFFFu) | ((tile0 & 0xFFFFu) << 16));
  g1[4] = (int)rfl(tile1 & 0xFFFFu);                 // tile_dim1 (tile_dim2=0)
  g1[5] = (int)rfl(stride0);                         // dim0_stride[31:0]
  g1[6] = 0;                                         // stride MSBs / dim1_stride
  g1[7] = 0;
  v4i_ gz4 = {0, 0, 0, 0};
  v8i_ gz8 = {0, 0, 0, 0, 0, 0, 0, 0};
  __builtin_amdgcn_tensor_load_to_lds(g0, g1, gz4, gz4, gz8, 0);
}

// ---------------------------------------------------------------------------
// Elementwise conversion: f32 -> bf16
// ---------------------------------------------------------------------------
__global__ void f32_to_bf16_kernel(const float* __restrict__ in,
                                   __bf16* __restrict__ out, int n) {
  int i = blockIdx.x * blockDim.x + threadIdx.x;
  if (i < n) out[i] = (__bf16)in[i];
}

// W[k,n] (KxN row-major, f32) -> Wt[n,k] (NxK row-major, bf16)
__global__ void transpose_to_bf16_kernel(const float* __restrict__ W,
                                         __bf16* __restrict__ Wt,
                                         int K, int N) {
  int i = blockIdx.x * blockDim.x + threadIdx.x;
  if (i >= K * N) return;
  int k = i / N, n = i % N;
  Wt[(size_t)n * K + k] = (__bf16)W[i];
}

// ---------------------------------------------------------------------------
// RoPE on the flat projected dim (reference applies it before head split),
// then scatter into head-major bf16 [B, nh, S, HD].
// ---------------------------------------------------------------------------
__global__ void rope_split_kernel(const float* __restrict__ proj, // [B,S,d]
                                  __bf16* __restrict__ out,       // [B,nh,S,HD]
                                  int d, int nh) {
  int halfd = d >> 1;
  int i = blockIdx.x * blockDim.x + threadIdx.x;
  if (i >= B_ * S_ * halfd) return;
  int j = i % halfd;
  int s = (i / halfd) % S_;
  int b = i / (halfd * S_);
  float inv = __expf(-logf(10000.0f) * (2.0f * (float)j) / (float)d);
  float ang = (float)s * inv;
  float sn, cs;
  __sincosf(ang, &sn, &cs);
  const float* p = proj + ((size_t)b * S_ + s) * d;
  float x1 = p[2 * j], x2 = p[2 * j + 1];
  float r1 = x1 * cs - x2 * sn;
  float r2 = x1 * sn + x2 * cs;
  int e1 = 2 * j, e2 = 2 * j + 1;
  out[(((size_t)b * nh + (e1 >> 6)) * S_ + s) * HD_ + (e1 & 63)] = (__bf16)r1;
  out[(((size_t)b * nh + (e2 >> 6)) * S_ + s) * HD_ + (e2 & 63)] = (__bf16)r2;
}

// V [B,S,HKV*HD] f32 -> Vt [B,HKV,HD,S] bf16
__global__ void v_transpose_kernel(const float* __restrict__ vf,
                                   __bf16* __restrict__ Vt) {
  int i = blockIdx.x * blockDim.x + threadIdx.x;
  if (i >= B_ * S_ * HKV_ * HD_) return;
  int e = i % (HKV_ * HD_);
  int s = (i / (HKV_ * HD_)) % S_;
  int b = i / (HKV_ * HD_ * S_);
  int g = e >> 6, hd = e & 63;
  Vt[(((size_t)b * HKV_ + g) * HD_ + hd) * S_ + s] = (__bf16)vf[i];
}

// ---------------------------------------------------------------------------
// Fragment helpers (contiguous k order used by this code, not the ISA table order; wave32)
//   A 16x32: lane<16 -> row=lane,    K = k0 + 0..15 (contiguous)
//            lane>=16 -> row=lane-16, K = k0 + 16..31 (contiguous)
//   B 32x16: col = lane%16, K = k0 + (lane<16 ? 0..15 : 16..31) (contiguous)
// ---------------------------------------------------------------------------
__device__ __forceinline__ v16bf make_a_frag(const __bf16* rowptr, int k0,
                                             int hf) {
  const __bf16* p0 = rowptr + k0 + hf * 16;
  v8bf a0 = *(const v8bf*)p0;
  v8bf a1 = *(const v8bf*)(p0 + 8);
  v16bf r;
#pragma unroll
  for (int i = 0; i < 8; ++i) { r[i] = a0[i]; r[i + 8] = a1[i]; }
  return r;
}

// ---------------------------------------------------------------------------
// bf16 WMMA GEMM, K fixed at 1024:  C[M,N] f32 = A[M,1024] x Bt[N,1024]^T
// 256 threads = 8 waves; each wave owns a 32x64 tile (8 WMMAs / k-step).
// Software double-buffered so next k-tile loads overlap current WMMAs.
// ---------------------------------------------------------------------------
#define GEMM_K 1024

__global__ __launch_bounds__(256) void gemm_bf16_kernel(
    const __bf16* __restrict__ A, const __bf16* __restrict__ Bt,
    float* __restrict__ C, int M, int N) {
  const int K = GEMM_K;
  int lane = threadIdx.x & 31;
  int wid  = threadIdx.x >> 5;
  int m0 = blockIdx.x * 256 + wid * 32;
  int n0 = blockIdx.y * 64;
  int hf  = (lane >> 4) & 1;
  int l16 = lane & 15;

  const __bf16* arow0 = A + (size_t)(m0 + l16) * K;
  const __bf16* arow1 = A + (size_t)(m0 + 16 + l16) * K;
  const __bf16* brow  = Bt + (size_t)(n0 + l16) * K;

  v8f acc[2][4];
#pragma unroll
  for (int mt = 0; mt < 2; ++mt)
#pragma unroll
    for (int nt = 0; nt < 4; ++nt) acc[mt][nt] = (v8f){};

  v16bf a_cur[2], b_cur[4];
  a_cur[0] = make_a_frag(arow0, 0, hf);
  a_cur[1] = make_a_frag(arow1, 0, hf);
#pragma unroll
  for (int nt = 0; nt < 4; ++nt)
    b_cur[nt] = *(const v16bf*)(brow + (size_t)nt * 16 * K + hf * 16);

#pragma unroll 4
  for (int k0 = 0; k0 < K; k0 += 32) {
    v16bf a_nxt[2], b_nxt[4];
    if (k0 + 32 < K) {
      int kn = k0 + 32;
      a_nxt[0] = make_a_frag(arow0, kn, hf);
      a_nxt[1] = make_a_frag(arow1, kn, hf);
#pragma unroll
      for (int nt = 0; nt < 4; ++nt)
        b_nxt[nt] = *(const v16bf*)(brow + (size_t)nt * 16 * K + kn + hf * 16);
      if (kn + 256 < K) {
        __builtin_prefetch(arow0 + kn + 256, 0, 1);   // global_prefetch_b8
        __builtin_prefetch(arow1 + kn + 256, 0, 1);
        __builtin_prefetch(brow + kn + 256, 0, 1);
      }
    }
#pragma unroll
    for (int mt = 0; mt < 2; ++mt)
#pragma unroll
      for (int nt = 0; nt < 4; ++nt)
        acc[mt][nt] = __builtin_amdgcn_wmma_f32_16x16x32_bf16(
            false, a_cur[mt], false, b_cur[nt], (short)0, acc[mt][nt], false,
            false);
#pragma unroll
    for (int mt = 0; mt < 2; ++mt) a_cur[mt] = a_nxt[mt];
#pragma unroll
    for (int nt = 0; nt < 4; ++nt) b_cur[nt] = b_nxt[nt];
  }

#pragma unroll
  for (int mt = 0; mt < 2; ++mt)
#pragma unroll
    for (int nt = 0; nt < 4; ++nt)
#pragma unroll
      for (int i = 0; i < 8; ++i) {
        int row = m0 + mt * 16 + i + hf * 8;
        C[(size_t)row * N + n0 + nt * 16 + l16] = acc[mt][nt][i];
      }
}

// ---------------------------------------------------------------------------
// Flash attention (GQA): 1 wave = 16 query rows x full HD.
// K/V tiles are DMA'd into LDS by the Tensor Data Mover (double-buffered,
// synchronized with s_wait_tensorcnt); P tile staged through LDS to rebuild
// the A-fragment layout. Block = 4 independent waves.
//
// Per-wave LDS slice (17408 B):
//   [0,8192)     K tiles, 2 x 4096 B  (32 kv rows x 64 bf16)
//   [8192,16384) V tiles, 2 x 4096 B  (64 hd rows x 32 bf16)
//   [16384,17408) P stage, 16 x 32 bf16
// ---------------------------------------------------------------------------
#define SLICE_B 17408

__global__ __launch_bounds__(128) void attn_kernel(
    const __bf16* __restrict__ Qbf,  // [B,HQ,S,HD]
    const __bf16* __restrict__ Kbf,  // [B,HKV,S,HD]
    const __bf16* __restrict__ Vt,   // [B,HKV,HD,S]
    __bf16* __restrict__ Obf) {      // [B,S,HQ*HD]
  extern __shared__ char smem_raw[];

  int lane = threadIdx.x & 31;
  int wid  = threadIdx.x >> 5;
  int w    = blockIdx.x * 4 + wid;
  int sblk = w & 127;               // S/16 = 128 blocks
  int h    = (w >> 7) & (HQ_ - 1);
  int b    = w >> 11;
  int g    = h >> 2;                // REP = 4
  int s0   = sblk * 16;

  const __bf16* Qb = Qbf + (((size_t)(b * HQ_ + h)) * S_ + s0) * HD_;
  const __bf16* Kb = Kbf + ((size_t)(b * HKV_ + g)) * S_ * HD_;
  const __bf16* Vb = Vt  + ((size_t)(b * HKV_ + g)) * HD_ * S_;

  int hf  = (lane >> 4) & 1;
  int l16 = lane & 15;

  unsigned slice = (unsigned)wid * SLICE_B;
  __bf16* pl = (__bf16*)(smem_raw + slice + 16384);

  // --- TDM issue helpers ----------------------------------------------------
  auto issue_k = [&](int buf, int t0) {
    // K tile: 32 rows x 64 bf16, contiguous in memory -> 1D copy of 2048 elems
    tdm_load_to_lds(slice + (unsigned)buf * 4096u, Kb + (size_t)t0 * HD_,
                    /*tile0=*/2048u, /*tile1=*/0u,
                    /*tens0=*/(unsigned)(S_ * HD_), /*tens1=*/1u,
                    /*stride0=*/(unsigned)(S_ * HD_));
  };
  auto issue_v = [&](int buf, int t0) {
    // V tile: 64 rows (hd) x 32 elems (t), row stride S_ -> 2D TDM
    tdm_load_to_lds(slice + 8192u + (unsigned)buf * 4096u, Vb + t0,
                    /*tile0=*/32u, /*tile1=*/64u,
                    /*tens0=*/(unsigned)S_, /*tens1=*/(unsigned)HD_,
                    /*stride0=*/(unsigned)S_);
  };

  // Q fragments for the two k-steps covering HD=64
  v16bf qa[2];
#pragma unroll
  for (int kk = 0; kk < 2; ++kk)
    qa[kk] = make_a_frag(Qb + (size_t)l16 * HD_, kk * 32, hf);

  v8f o[4];
#pragma unroll
  for (int nt = 0; nt < 4; ++nt) o[nt] = (v8f){};

  float mrow[16], lrow[16];
#pragma unroll
  for (int r = 0; r < 16; ++r) { mrow[r] = -1e30f; lrow[r] = 0.0f; }

  issue_k(0, 0);
  issue_v(0, 0);

  for (int t0 = 0; t0 < S_; t0 += 32) {
    int cur = (t0 >> 5) & 1;
    if (t0 + 32 < S_) {
      issue_k(cur ^ 1, t0 + 32);
      issue_v(cur ^ 1, t0 + 32);
      __builtin_amdgcn_s_wait_tensorcnt((short)2);  // current tile landed
    } else {
      __builtin_amdgcn_s_wait_tensorcnt((short)0);
    }
    const __bf16* kl = (const __bf16*)(smem_raw + slice + cur * 4096);
    const __bf16* vl = (const __bf16*)(smem_raw + slice + 8192 + cur * 4096);

    // ---- scores: 16 x 32 tile = two 16x16 C tiles --------------------------
    v8f st[2];
#pragma unroll
    for (int ct = 0; ct < 2; ++ct) {
      v8f acc = (v8f){};
#pragma unroll
      for (int kk = 0; kk < 2; ++kk) {
        const __bf16* bp = kl + (ct * 16 + l16) * HD_ + kk * 32 + hf * 16;
        v16bf bf = *(const v16bf*)bp;
        acc = __builtin_amdgcn_wmma_f32_16x16x32_bf16(
            false, qa[kk], false, bf, (short)0, acc, false, false);
      }
#pragma unroll
      for (int i = 0; i < 8; ++i) acc[i] *= 0.125f;  // 1/sqrt(64)
      st[ct] = acc;
    }

    // ---- online softmax row stats (replicated across lanes) ----------------
    float newm[16], alpha[16], tsum[16];
#pragma unroll
    for (int i = 0; i < 8; ++i) {
      float v = fmaxf(st[0][i], st[1][i]);
      v = fmaxf(v, __shfl_xor(v, 1, 32));
      v = fmaxf(v, __shfl_xor(v, 2, 32));
      v = fmaxf(v, __shfl_xor(v, 4, 32));
      v = fmaxf(v, __shfl_xor(v, 8, 32));
      float ov = __shfl_xor(v, 16, 32);
      float lo = (lane < 16) ? v : ov;
      float hi = (lane < 16) ? ov : v;
      newm[i]     = fmaxf(mrow[i], lo);
      newm[i + 8] = fmaxf(mrow[i + 8], hi);
    }
#pragma unroll
    for (int r = 0; r < 16; ++r) alpha[r] = __expf(mrow[r] - newm[r]);

    // ---- exponentiate, row sums, stage P tile in LDS -----------------------
#pragma unroll
    for (int i = 0; i < 8; ++i) {
      float mr = (lane & 16) ? newm[i + 8] : newm[i];
      float p0 = __expf(st[0][i] - mr);
      float p1 = __expf(st[1][i] - mr);
      int row = i + (hf << 3);
      pl[row * 32 + l16]      = (__bf16)p0;
      pl[row * 32 + 16 + l16] = (__bf16)p1;
      float v = p0 + p1;
      v += __shfl_xor(v, 1, 32);
      v += __shfl_xor(v, 2, 32);
      v += __shfl_xor(v, 4, 32);
      v += __shfl_xor(v, 8, 32);
      float ov = __shfl_xor(v, 16, 32);
      tsum[i]     = (lane < 16) ? v : ov;
      tsum[i + 8] = (lane < 16) ? ov : v;
    }
#pragma unroll
    for (int r = 0; r < 16; ++r) {
      lrow[r] = lrow[r] * alpha[r] + tsum[r];
      mrow[r] = newm[r];
    }

    // ---- rescale accumulators ----------------------------------------------
#pragma unroll
    for (int nt = 0; nt < 4; ++nt)
#pragma unroll
      for (int i = 0; i < 8; ++i) {
        float a = (lane & 16) ? alpha[i + 8] : alpha[i];
        o[nt][i] *= a;
      }

    // wave-internal LDS RAW on the P tile: drain DScnt before re-reading
    asm volatile("s_wait_dscnt 0x0" ::: "memory");

    v16bf pa = make_a_frag(pl + l16 * 32, 0, hf);

    // ---- PV: O += P (16x32) x V (32x64) ------------------------------------
#pragma unroll
    for (int nt = 0; nt < 4; ++nt) {
      const __bf16* bp = vl + (nt * 16 + l16) * 32 + hf * 16;
      v16bf bf = *(const v16bf*)bp;
      o[nt] = __builtin_amdgcn_wmma_f32_16x16x32_bf16(
          false, pa, false, bf, (short)0, o[nt], false, false);
    }
  }

  // ---- epilogue: normalize by row sums, write bf16 [B,S,HQ*HD] -------------
  float linv[16];
#pragma unroll
  for (int r = 0; r < 16; ++r) linv[r] = 1.0f / lrow[r];

#pragma unroll
  for (int nt = 0; nt < 4; ++nt)
#pragma unroll
    for (int i = 0; i < 8; ++i) {
      int row = i + (hf << 3);
      float li = (lane & 16) ? linv[i + 8] : linv[i];
      size_t off = (((size_t)b * S_) + (s0 + row)) * (size_t)D_ +
                   (size_t)h * HD_ + nt * 16 + l16;
      Obf[off] = (__bf16)(o[nt][i] * li);
    }
}

// ---------------------------------------------------------------------------
// Host-side launcher
// ---------------------------------------------------------------------------
extern "C" void kernel_launch(void* const* d_in, const int* in_sizes, int n_in,
                              void* d_out, int out_size, void* d_ws,
                              size_t ws_size, hipStream_t stream) {
  (void)in_sizes; (void)n_in; (void)out_size; (void)ws_size;
  const float* x  = (const float*)d_in[0];  // [B,S,D]
  const float* wq = (const float*)d_in[1];  // [D, HQ*HD]
  const float* wk = (const float*)d_in[2];  // [D, HKV*HD]
  const float* wv = (const float*)d_in[3];  // [D, HKV*HD]
  const float* wo = (const float*)d_in[4];  // [D, D]
  float* out = (float*)d_out;               // [B,S,D]

  const size_t M  = (size_t)B_ * S_;        // 8192
  const size_t NK = (size_t)HKV_ * HD_;     // 256

  char* ws = (char*)d_ws;
  size_t off = 0;
  auto take = [&](size_t bytes) { char* p = ws + off; off += bytes; return p; };

  __bf16* xbf = (__bf16*)take(M * D_ * 2);            // 16 MB
  __bf16* wqT = (__bf16*)take((size_t)D_ * D_ * 2);   //  2 MB  [N=1024,K=1024]
  __bf16* wkT = (__bf16*)take((size_t)D_ * NK * 2);   //  0.5MB [N=256, K=1024]
  __bf16* wvT = (__bf16*)take((size_t)D_ * NK * 2);
  __bf16* woT = (__bf16*)take((size_t)D_ * D_ * 2);
  float*  qf  = (float*)take(M * D_ * 4);             // 32 MB
  float*  kf  = (float*)take(M * NK * 4);             //  8 MB
  float*  vf  = (float*)take(M * NK * 4);
  __bf16* Qbf = (__bf16*)take(M * D_ * 2);            // [B,HQ,S,HD]
  __bf16* Kbf = (__bf16*)take(M * NK * 2);            // [B,HKV,S,HD]
  __bf16* VtT = (__bf16*)take(M * NK * 2);            // [B,HKV,HD,S]
  __bf16* Obf = (__bf16*)take(M * D_ * 2);            // [B,S,D]

  const int T = 256;

  // 1) convert inputs / transpose weights to bf16
  f32_to_bf16_kernel<<<(int)((M * D_ + T - 1) / T), T, 0, stream>>>(
      x, xbf, (int)(M * D_));
  transpose_to_bf16_kernel<<<(int)(((size_t)D_ * D_ + T - 1) / T), T, 0,
                             stream>>>(wq, wqT, D_, D_);
  transpose_to_bf16_kernel<<<(int)(((size_t)D_ * NK + T - 1) / T), T, 0,
                             stream>>>(wk, wkT, D_, (int)NK);
  transpose_to_bf16_kernel<<<(int)(((size_t)D_ * NK + T - 1) / T), T, 0,
                             stream>>>(wv, wvT, D_, (int)NK);
  transpose_to_bf16_kernel<<<(int)(((size_t)D_ * D_ + T - 1) / T), T, 0,
                             stream>>>(wo, woT, D_, D_);

  // 2) QKV projections (WMMA GEMM, K=1024)
  gemm_bf16_kernel<<<dim3((int)(M / 256), D_ / 64), 256, 0, stream>>>(
      xbf, wqT, qf, (int)M, D_);
  gemm_bf16_kernel<<<dim3((int)(M / 256), (int)(NK / 64)), 256, 0, stream>>>(
      xbf, wkT, kf, (int)M, (int)NK);
  gemm_bf16_kernel<<<dim3((int)(M / 256), (int)(NK / 64)), 256, 0, stream>>>(
      xbf, wvT, vf, (int)M, (int)NK);

  // 3) RoPE (flat dim, as in reference) + head split + V transpose
  rope_split_kernel<<<(int)((M * (D_ / 2) + T - 1) / T), T, 0, stream>>>(
      qf, Qbf, D_, HQ_);
  rope_split_kernel<<<(int)((M * (NK / 2) + T - 1) / T), T, 0, stream>>>(
      kf, Kbf, (int)NK, HKV_);
  v_transpose_kernel<<<(int)((M * NK + T - 1) / T), T, 0, stream>>>(vf, VtT);

  // 4) flash attention: TDM-staged K/V + WMMA QK^T/PV + online softmax
  attn_kernel<<<(B_ * HQ_ * (S_ / 16)) / 4, 128, 4 * SLICE_B, stream>>>(
      Qbf, Kbf, VtT, Obf);

  // 5) output projection (WMMA GEMM) -> f32 d_out
  gemm_bf16_kernel<<<dim3((int)(M / 256), D_ / 64), 256, 0, stream>>>(
      Obf, woT, out, (int)M, D_);
}
